// VoxelNetBackbone_36240934044336
// MI455X (gfx1250) — matrix operand order repaired and checked by reading, not compiled, not run on hardware
//
#include <hip/hip_runtime.h>
#include <hip/hip_bf16.h>

typedef __attribute__((ext_vector_type(16))) __bf16 v16bf;
typedef __attribute__((ext_vector_type(8)))  float  v8f;

#define GD 50
#define GD2 25
#define VDIM 128
#define NB 4
#define NVOX 8192
#define NPTS 32

// ---------------- helpers ----------------

static __device__ inline unsigned short f2bf(float f) {
  union { float f; unsigned int u; } x; x.f = f;
  unsigned int u = x.u;
  u += 0x7fffu + ((u >> 16) & 1u);       // round-to-nearest-even
  return (unsigned short)(u >> 16);
}

static __device__ inline v8f wmma_bf16(v16bf a, v16bf b, v8f c) {
  return __builtin_amdgcn_wmma_f32_16x16x32_bf16(false, a, false, b,
                                                 (short)0, c, false, false);
}

// A fragment (16x32 bf16). `base` points at the 32-channel window for this
// lane's row (4-byte aligned). Layout: lane half h, element j:
//   j in [0,16): K = 16h + j -> dwords 8h .. 8h+7
static __device__ inline v16bf load_A(const unsigned short* base, int half) {
  union { v16bf v; unsigned int u[8]; } a;
  const unsigned int* r = (const unsigned int*)base;
  int d0 = half * 8;
  int d1 = half * 8 + 4;
#pragma unroll
  for (int i = 0; i < 4; ++i) { a.u[i] = r[d0 + i]; a.u[4 + i] = r[d1 + i]; }
  return a.v;
}

// B fragment (32x16 bf16) from memory laid out [n][K] (K contiguous).
// `p` = &buf[n*Kdim + kbase], kbase = kstep*32 + half*16, 4-byte aligned.
static __device__ inline v16bf load_B(const unsigned short* p) {
  union { v16bf v; unsigned int u[8]; } b;
  const unsigned int* r = (const unsigned int*)p;
#pragma unroll
  for (int i = 0; i < 8; ++i) b.u[i] = r[i];
  return b.v;
}

// ---------------- one-shot packing kernels ----------------

__global__ void zero_kernel(unsigned int* p, long n) {
  long i = (long)blockIdx.x * blockDim.x + threadIdx.x;
  if (i < n) p[i] = 0u;
}

// W[K][N] fp32 (row-major) -> out[n*Kpad + k] bf16, zero-padded past kvalid
__global__ void pack_w_kernel(const float* __restrict__ W, int N, int Kpad,
                              int kvalid, unsigned short* __restrict__ out) {
  int i = blockIdx.x * blockDim.x + threadIdx.x;
  if (i >= N * Kpad) return;
  int n = i / Kpad, k = i % Kpad;
  out[i] = (k < kvalid) ? f2bf(W[k * N + n]) : (unsigned short)0;
}

// k[(o*Cin+i)*27 + tap] fp32 -> out[tap*Cout*Cin + o*Cin + i] bf16
__global__ void pack_k_kernel(const float* __restrict__ k, int Cout, int Cin,
                              unsigned short* __restrict__ out) {
  long idx = (long)blockIdx.x * blockDim.x + threadIdx.x;
  long per = (long)Cout * Cin;
  if (idx >= 27 * per) return;
  int tap = (int)(idx / per);
  long rem = idx % per;
  out[idx] = f2bf(k[rem * 27 + tap]);
}

// ---------------- VFE: per-voxel MLP + maxpool + scatter ----------------
__global__ __launch_bounds__(128) void vfe_kernel(
    const float* __restrict__ vf, const int* __restrict__ coords,
    const unsigned short* __restrict__ W1p, const float* __restrict__ b1,
    const unsigned short* __restrict__ W2p, const float* __restrict__ b2,
    const unsigned short* __restrict__ W3p, const float* __restrict__ b3,
    unsigned short* __restrict__ grid) {
  __shared__ unsigned short Xs[4][32 * 32];    // 8 KB
  __shared__ unsigned short H1s[4][32 * 64];   // 16 KB
  __shared__ unsigned short H2s[4][32 * 128];  // 32 KB
  const int tid = threadIdx.x;
  const int wave = tid >> 5;
  const int lane = tid & 31;
  const int half = lane >> 4;
  const int lrow = lane & 15;
  const int vox = blockIdx.x * 4 + wave;       // 0..32767
  const int b = vox >> 13;
  const int n = vox & (NVOX - 1);

  unsigned short* X  = Xs[wave];
  unsigned short* H1 = H1s[wave];
  unsigned short* H2 = H2s[wave];

  // stage X (32 pts x 32 ch, zero padded past C=8) into LDS as bf16
  for (int e = lane; e < 32 * 32; e += 32) X[e] = 0;
  const float* xp = vf + (((long)(b * NVOX + n)) * NPTS + lane) * 8;
#pragma unroll
  for (int c = 0; c < 8; ++c) X[lane * 32 + c] = f2bf(xp[c]);
  __syncthreads();

  // ---- layer 1: (32x8)@(8x64) + b1, ReLU ----
  {
    v16bf a0 = load_A(X + (lrow) * 32, half);
    v16bf a1 = load_A(X + (16 + lrow) * 32, half);
#pragma unroll
    for (int t = 0; t < 4; ++t) {
      float bv = b1[t * 16 + lrow];
      v8f c0 = {bv, bv, bv, bv, bv, bv, bv, bv};
      v8f c1 = c0;
      v16bf bb = load_B(W1p + (t * 16 + lrow) * 32 + half * 16);
      c0 = wmma_bf16(a0, bb, c0);
      c1 = wmma_bf16(a1, bb, c1);
      int col = t * 16 + lrow;
#pragma unroll
      for (int r = 0; r < 8; ++r) {
        H1[(r + 8 * half) * 64 + col]      = f2bf(fmaxf(c0[r], 0.0f));
        H1[(16 + r + 8 * half) * 64 + col] = f2bf(fmaxf(c1[r], 0.0f));
      }
    }
  }
  __syncthreads();

  // ---- layer 2: (32x64)@(64x128) + b2, ReLU ----
  {
    v8f d0[8], d1[8];
#pragma unroll
    for (int t = 0; t < 8; ++t) {
      float bv = b2[t * 16 + lrow];
#pragma unroll
      for (int r = 0; r < 8; ++r) { d0[t][r] = bv; d1[t][r] = bv; }
    }
#pragma unroll
    for (int ks = 0; ks < 2; ++ks) {
      v16bf a0 = load_A(H1 + (lrow) * 64 + ks * 32, half);
      v16bf a1 = load_A(H1 + (16 + lrow) * 64 + ks * 32, half);
#pragma unroll
      for (int t = 0; t < 8; ++t) {
        v16bf bb = load_B(W2p + (t * 16 + lrow) * 64 + ks * 32 + half * 16);
        d0[t] = wmma_bf16(a0, bb, d0[t]);
        d1[t] = wmma_bf16(a1, bb, d1[t]);
      }
    }
#pragma unroll
    for (int t = 0; t < 8; ++t) {
      int col = t * 16 + lrow;
#pragma unroll
      for (int r = 0; r < 8; ++r) {
        H2[(r + 8 * half) * 128 + col]      = f2bf(fmaxf(d0[t][r], 0.0f));
        H2[(16 + r + 8 * half) * 128 + col] = f2bf(fmaxf(d1[t][r], 0.0f));
      }
    }
  }
  __syncthreads();

  // ---- layer 3: (32x128)@(128x128) + b3, then max over 32 rows, scatter ----
  const int vx = coords[(long)(b * NVOX + n) * 3 + 0];
  const int vy = coords[(long)(b * NVOX + n) * 3 + 1];
  const int vz = coords[(long)(b * NVOX + n) * 3 + 2];
  unsigned short* gout =
      grid + ((((long)(b * GD + vx)) * GD + vy) * GD + vz) * VDIM;
  {
    v8f e0[8], e1[8];
#pragma unroll
    for (int t = 0; t < 8; ++t) {
      float bv = b3[t * 16 + lrow];
#pragma unroll
      for (int r = 0; r < 8; ++r) { e0[t][r] = bv; e1[t][r] = bv; }
    }
#pragma unroll
    for (int ks = 0; ks < 4; ++ks) {
      v16bf a0 = load_A(H2 + (lrow) * 128 + ks * 32, half);
      v16bf a1 = load_A(H2 + (16 + lrow) * 128 + ks * 32, half);
#pragma unroll
      for (int t = 0; t < 8; ++t) {
        v16bf bb = load_B(W3p + (t * 16 + lrow) * 128 + ks * 32 + half * 16);
        e0[t] = wmma_bf16(a0, bb, e0[t]);
        e1[t] = wmma_bf16(a1, bb, e1[t]);
      }
    }
#pragma unroll
    for (int t = 0; t < 8; ++t) {
      float m = -3.4e38f;
#pragma unroll
      for (int r = 0; r < 8; ++r) { m = fmaxf(m, e0[t][r]); m = fmaxf(m, e1[t][r]); }
      m = fmaxf(m, __shfl_xor(m, 16, 32));  // lanes l / l+16 share column
      if (half == 0) gout[t * 16 + lrow] = f2bf(m);
    }
  }
}

// ---------------- conv1: 3^3, 128->64, stride 1, implicit GEMM ----------------
__global__ __launch_bounds__(128) void conv1_kernel(
    const unsigned short* __restrict__ grid,
    const unsigned short* __restrict__ k1p,
    float* __restrict__ raw1, float* s1, float* q1) {
  __shared__ unsigned short Ain[52 * 128];  // w = -1..50 input row, 13 KB
  __shared__ unsigned short Bw[64 * 128];   // tap weights [o][i] bf16, 16 KB
  const int tid = threadIdx.x;
  const int wave = tid >> 5;
  const int lane = tid & 31;
  const int half = lane >> 4;
  const int lrow = lane & 15;
  const int blk = blockIdx.x;                 // over B*50*50
  const int b = blk / 2500;
  const int rem = blk % 2500;
  const int x = rem / GD;
  const int y = rem % GD;
  const int w0 = wave * 16;

  v8f acc[4];
#pragma unroll
  for (int t = 0; t < 4; ++t)
#pragma unroll
    for (int r = 0; r < 8; ++r) acc[t][r] = 0.0f;

  for (int g = 0; g < 9; ++g) {               // (dd, dh) groups
    const int dd = g / 3, dh = g % 3;
    const int xi = x + dd - 1, yi = y + dh - 1;
    if ((unsigned)xi >= (unsigned)GD || (unsigned)yi >= (unsigned)GD) continue;  // block-uniform

    // stage input row (50 positions x 128 ch) with zero pads at w=-1, 50
    {
      unsigned int* A32 = (unsigned int*)Ain;
      const unsigned int* src = (const unsigned int*)(
          grid + ((((long)(b * GD + xi)) * GD + yi) * GD) * VDIM);
      if (tid < 64) { A32[tid] = 0u; A32[51 * 64 + tid] = 0u; }
      for (int e = tid; e < 3200; e += 128) A32[64 + e] = src[e];
    }

    for (int dw = 0; dw < 3; ++dw) {
      const int tap = g * 3 + dw;             // dd*9 + dh*3 + dw
      {
        unsigned int* B32 = (unsigned int*)Bw;
        const unsigned int* ksrc = (const unsigned int*)(k1p + (long)tap * 8192);
        for (int e = tid; e < 4096; e += 128) B32[e] = ksrc[e];
      }
      __syncthreads();                        // Ain + Bw visible

      int arow = w0 + lrow + dw;              // = wi + 1
      if (arow > 51) arow = 51;               // junk rows never stored
      const unsigned short* abase = Ain + arow * 128;
#pragma unroll
      for (int ks = 0; ks < 4; ++ks) {
        v16bf a = load_A(abase + ks * 32, half);
#pragma unroll
        for (int t = 0; t < 4; ++t) {
          const unsigned short* bp =
              Bw + (t * 16 + lrow) * 128 + ks * 32 + half * 16;
          acc[t] = wmma_bf16(a, load_B(bp), acc[t]);
        }
      }
      __syncthreads();                        // protect Bw/Ain restage
    }
  }

  const long obase = ((((long)(b * GD + x)) * GD + y) * GD) * 64;
#pragma unroll
  for (int t = 0; t < 4; ++t) {
    const int o = t * 16 + lrow;
    float ls = 0.0f, lq = 0.0f;
#pragma unroll
    for (int r = 0; r < 8; ++r) {
      int wout = w0 + r + 8 * half;
      float v = acc[t][r];
      if (wout < GD) { raw1[obase + (long)wout * 64 + o] = v; ls += v; lq += v * v; }
    }
    ls += __shfl_xor(ls, 16, 32);
    lq += __shfl_xor(lq, 16, 32);
    if (half == 0) { atomicAdd(&s1[o], ls); atomicAdd(&q1[o], lq); }
  }
}

// ---------------- BN1 + ReLU -> bf16 activations ----------------
__global__ void bn1_kernel(const float* __restrict__ raw1, const float* s1,
                           const float* q1, const float* __restrict__ g,
                           const float* __restrict__ be,
                           unsigned short* __restrict__ act1, long n) {
  long i = (long)blockIdx.x * blockDim.x + threadIdx.x;
  if (i >= n) return;
  int o = (int)(i & 63);
  const float cnt = 500000.0f;  // 4*50^3
  float mean = s1[o] / cnt;
  float var = q1[o] / cnt - mean * mean;
  float v = (raw1[i] - mean) * rsqrtf(var + 1e-5f) * g[o] + be[o];
  act1[i] = f2bf(fmaxf(v, 0.0f));
}

// ---------------- conv2: 3^3, 64->128, stride 2 ----------------
__global__ __launch_bounds__(64) void conv2_kernel(
    const unsigned short* __restrict__ act1,
    const unsigned short* __restrict__ k2p,
    float* __restrict__ raw2, float* s2, float* q2) {
  __shared__ unsigned short Ain[52 * 64];   // 6.5 KB
  __shared__ unsigned short Bw[128 * 64];   // [o][i] bf16, 16 KB
  const int tid = threadIdx.x;
  const int wave = tid >> 5;
  const int lane = tid & 31;
  const int half = lane >> 4;
  const int lrow = lane & 15;
  const int blk = blockIdx.x;              // over B*25*25
  const int b = blk / 625;
  const int rem = blk % 625;
  const int x2 = rem / GD2;
  const int y2 = rem % GD2;
  const int w0 = wave * 16;

  v8f acc[8];
#pragma unroll
  for (int t = 0; t < 8; ++t)
#pragma unroll
    for (int r = 0; r < 8; ++r) acc[t][r] = 0.0f;

  for (int g = 0; g < 9; ++g) {
    const int dd = g / 3, dh = g % 3;
    const int xi = 2 * x2 + dd - 1, yi = 2 * y2 + dh - 1;
    if ((unsigned)xi >= (unsigned)GD || (unsigned)yi >= (unsigned)GD) continue;

    {
      unsigned int* A32 = (unsigned int*)Ain;
      const unsigned int* src = (const unsigned int*)(
          act1 + ((((long)(b * GD + xi)) * GD + yi) * GD) * 64);
      if (tid < 32) { A32[tid] = 0u; A32[51 * 32 + tid] = 0u; }
      for (int e = tid; e < 1600; e += 64) A32[32 + e] = src[e];
    }

    for (int dw = 0; dw < 3; ++dw) {
      const int tap = g * 3 + dw;
      {
        unsigned int* B32 = (unsigned int*)Bw;
        const unsigned int* ksrc = (const unsigned int*)(k2p + (long)tap * 8192);
        for (int e = tid; e < 4096; e += 64) B32[e] = ksrc[e];
      }
      __syncthreads();

      int arow = 2 * (w0 + lrow) + dw;       // = wi + 1
      if (arow > 51) arow = 51;              // junk rows never stored
      const unsigned short* abase = Ain + arow * 64;
#pragma unroll
      for (int ks = 0; ks < 2; ++ks) {
        v16bf a = load_A(abase + ks * 32, half);
#pragma unroll
        for (int t = 0; t < 8; ++t) {
          const unsigned short* bp =
              Bw + (t * 16 + lrow) * 64 + ks * 32 + half * 16;
          acc[t] = wmma_bf16(a, load_B(bp), acc[t]);
        }
      }
      __syncthreads();
    }
  }

  const long obase = ((((long)(b * GD2 + x2)) * GD2 + y2) * GD2) * 128;
#pragma unroll
  for (int t = 0; t < 8; ++t) {
    const int o = t * 16 + lrow;
    float ls = 0.0f, lq = 0.0f;
#pragma unroll
    for (int r = 0; r < 8; ++r) {
      int wo = w0 + r + 8 * half;
      float v = acc[t][r];
      if (wo < GD2) { raw2[obase + (long)wo * 128 + o] = v; ls += v; lq += v * v; }
    }
    ls += __shfl_xor(ls, 16, 32);
    lq += __shfl_xor(lq, 16, 32);
    if (half == 0) { atomicAdd(&s2[o], ls); atomicAdd(&q2[o], lq); }
  }
}

// ---------------- BN2 + ReLU -> NCDHW fp32 output ----------------
__global__ void bn2_kernel(const float* __restrict__ raw2, const float* s2,
                           const float* q2, const float* __restrict__ g,
                           const float* __restrict__ be,
                           float* __restrict__ out, long n) {
  long i = (long)blockIdx.x * blockDim.x + threadIdx.x;
  if (i >= n) return;
  int w = (int)(i % GD2); long r = i / GD2;
  int y = (int)(r % GD2); r /= GD2;
  int x = (int)(r % GD2); r /= GD2;
  int o = (int)(r % 128);
  int b = (int)(r / 128);
  const float cnt = 62500.0f;  // 4*25^3
  float mean = s2[o] / cnt;
  float var = q2[o] / cnt - mean * mean;
  float raw = raw2[((((long)(b * GD2 + x)) * GD2 + y) * GD2 + w) * 128 + o];
  float v = (raw - mean) * rsqrtf(var + 1e-5f) * g[o] + be[o];
  out[i] = fmaxf(v, 0.0f);
}

// ---------------- launch ----------------
extern "C" void kernel_launch(void* const* d_in, const int* in_sizes, int n_in,
                              void* d_out, int out_size, void* d_ws, size_t ws_size,
                              hipStream_t stream) {
  (void)in_sizes; (void)n_in; (void)out_size; (void)ws_size;
  const float* vf     = (const float*)d_in[0];
  const int*   coords = (const int*)d_in[1];
  const float* W1 = (const float*)d_in[2];
  const float* b1 = (const float*)d_in[3];
  const float* W2 = (const float*)d_in[4];
  const float* b2 = (const float*)d_in[5];
  const float* W3 = (const float*)d_in[6];
  const float* b3 = (const float*)d_in[7];
  const float* k1 = (const float*)d_in[8];
  const float* g1 = (const float*)d_in[9];
  const float* be1 = (const float*)d_in[10];
  const float* k2 = (const float*)d_in[11];
  const float* g2 = (const float*)d_in[12];
  const float* be2 = (const float*)d_in[13];
  float* out = (float*)d_out;

  char* ws = (char*)d_ws;
  const long GRID_ELEMS = (long)NB * GD * GD * GD * VDIM;    // 64M bf16
  const long RAW1_ELEMS = (long)NB * GD * GD * GD * 64;      // 32M f32
  const long RAW2_ELEMS = (long)NB * GD2 * GD2 * GD2 * 128;  // 8M f32

  unsigned short* grid = (unsigned short*)(ws);
  float* raw1          = (float*)(ws + 128000000L);
  unsigned short* act1 = (unsigned short*)(ws + 256000000L);
  float* raw2          = (float*)(ws + 320000000L);
  float* stats         = (float*)(ws + 352000000L);
  float* s1 = stats;        // 64
  float* q1 = stats + 64;   // 64
  float* s2 = stats + 128;  // 128
  float* q2 = stats + 256;  // 128
  unsigned short* W1p = (unsigned short*)(ws + 352002048L);  // 64*32
  unsigned short* W2p = (unsigned short*)(ws + 352006144L);  // 128*64
  unsigned short* W3p = (unsigned short*)(ws + 352022528L);  // 128*128
  unsigned short* k1p = (unsigned short*)(ws + 352055296L);  // 27*64*128
  unsigned short* k2p = (unsigned short*)(ws + 352497664L);  // 27*128*64

  // zero grid (as dwords) and stats
  {
    long ndw = GRID_ELEMS / 2;  // 32M dwords
    zero_kernel<<<(int)((ndw + 255) / 256), 256, 0, stream>>>((unsigned int*)grid, ndw);
    zero_kernel<<<2, 256, 0, stream>>>((unsigned int*)stats, 384);
  }

  // one-shot weight packing (fp32 -> bf16, fragment-friendly layouts)
  pack_w_kernel<<<(64 * 32 + 255) / 256, 256, 0, stream>>>(W1, 64, 32, 8, W1p);
  pack_w_kernel<<<(128 * 64 + 255) / 256, 256, 0, stream>>>(W2, 128, 64, 64, W2p);
  pack_w_kernel<<<(128 * 128 + 255) / 256, 256, 0, stream>>>(W3, 128, 128, 128, W3p);
  pack_k_kernel<<<(27 * 64 * 128 + 255) / 256, 256, 0, stream>>>(k1, 64, 128, k1p);
  pack_k_kernel<<<(27 * 128 * 64 + 255) / 256, 256, 0, stream>>>(k2, 128, 64, k2p);

  // VFE + scatter: 32768 voxels, 1 wave each, 4 waves/block
  vfe_kernel<<<(NB * NVOX) / 4, 128, 0, stream>>>(vf, coords, W1p, b1, W2p, b2,
                                                  W3p, b3, grid);

  // conv1 + stats
  conv1_kernel<<<NB * GD * GD, 128, 0, stream>>>(grid, k1p, raw1, s1, q1);

  // BN1 + ReLU -> bf16
  bn1_kernel<<<(int)((RAW1_ELEMS + 255) / 256), 256, 0, stream>>>(
      raw1, s1, q1, g1, be1, act1, RAW1_ELEMS);

  // conv2 + stats
  conv2_kernel<<<NB * GD2 * GD2, 64, 0, stream>>>(act1, k2p, raw2, s2, q2);

  // BN2 + ReLU -> NCDHW fp32 output
  bn2_kernel<<<(int)((RAW2_ELEMS + 255) / 256), 256, 0, stream>>>(
      raw2, s2, q2, g2, be2, out, RAW2_ELEMS);
}
